// Scale_Unified_Attention_82454782149125
// MI455X (gfx1250) — matrix operand order repaired and checked by reading, not compiled, not run on hardware
//
#include <hip/hip_runtime.h>
#include <hip/hip_bf16.h>
#include <cmath>

// ---------------- CDNA5 WMMA plumbing (wave32, 16x16x32 f16 -> f32) ----------------
typedef _Float16 h16 __attribute__((ext_vector_type(16)));
typedef float    f8  __attribute__((ext_vector_type(8)));

#define DEV __device__ __forceinline__

DEV f8 wmma_f16(h16 a, h16 b, f8 c) {
  // (neg_a, A, neg_b, B, c_mod, C, reuse_a, reuse_b)
  return __builtin_amdgcn_wmma_f32_16x16x32_f16(false, a, false, b, (short)0, c, false, false);
}

// A fragment: 16x32 (MxK) f16, row-major source base[m*ld + k]
// Layout: lane L -> M = L&15; VGPR j holds K = (L>>4)*16 + 2j (+1)
DEV h16 load_a(const _Float16* base, int ld, int lane) {
  const int m = lane & 15, hh = lane >> 4;
  h16 f;
#pragma unroll
  for (int j = 0; j < 8; ++j) {
    const int k = (hh << 4) + (j << 1);
    f[2*j]   = base[m*ld + k];
    f[2*j+1] = base[m*ld + k + 1];
  }
  return f;
}

// B fragment: 32x16 (KxN) f16, row-major source base[k*ld + n]
// Layout: lane L -> N = L&15; VGPR j holds K = (L>>4)*16 + 2j (+1)
DEV h16 load_b(const _Float16* base, int ld, int lane) {
  const int n = lane & 15, hh = lane >> 4;
  h16 f;
#pragma unroll
  for (int j = 0; j < 8; ++j) {
    const int k = (hh << 4) + (j << 1);
    f[2*j]   = base[k*ld + n];
    f[2*j+1] = base[(k+1)*ld + n];
  }
  return f;
}

// B fragment from transposed source: B[k][n] = base[n*ld + k]
DEV h16 load_bT(const _Float16* base, int ld, int lane) {
  const int n = lane & 15, hh = lane >> 4;
  h16 f;
#pragma unroll
  for (int j = 0; j < 8; ++j) {
    const int k = (hh << 4) + (j << 1);
    f[2*j]   = base[n*ld + k];
    f[2*j+1] = base[n*ld + k + 1];
  }
  return f;
}

// ---------------- misc helpers ----------------
__global__ void cvt_f32_f16(const float* __restrict__ src, _Float16* __restrict__ dst, int n) {
  int i = blockIdx.x * 256 + threadIdx.x;
  if (i < n) dst[i] = (_Float16)src[i];
}

DEV int seg85(int v) { return v < 64 ? 0 : (v < 80 ? 1 : (v < 84 ? 2 : 3)); }
DEV int oh85(int s)  { return s == 0 ? 0 : (s == 1 ? 1 : (s == 2 ? 5 : 21)); }
DEV int ow85(int s)  { return s == 0 ? 21 : (s == 1 ? 5 : (s == 2 ? 1 : 0)); }

// ================== Stage 1: window partition + LN + sq proj + qkv proj ==================
template<int CDIM, int SQi, int WSi, int OFFi>
__global__ __launch_bounds__(32)
void qkv_kernel(const float* __restrict__ x,
                const float* __restrict__ gamma, const float* __restrict__ beta,
                const _Float16* __restrict__ sqw, const float* __restrict__ sqb,
                const _Float16* __restrict__ qkvw, const float* __restrict__ qkvb,
                _Float16* __restrict__ Q, _Float16* __restrict__ K, _Float16* __restrict__ V) {
  __shared__ _Float16 Ax[16 * CDIM];  // LN'ed tokens, f16
  __shared__ _Float16 Y[16 * 96];     // after sq proj
  const int lane = threadIdx.x;
  const int tg0 = blockIdx.x * 16;
  const int HW = WSi * 16;

  if (lane < 16) {
    const int tg = tg0 + lane;
    const int wi = tg / SQi, t = tg % SQi;
    const int b = wi >> 8, wy = (wi >> 4) & 15, wx = wi & 15;
    const int ty = t / WSi, tx = t % WSi;
    const int row = wy * WSi + ty, col = wx * WSi + tx;
    const float* xr = x + ((size_t)(b * HW + row) * HW + col) * CDIM;
    float s = 0.f, s2 = 0.f;
    for (int c = 0; c < CDIM; ++c) { float v = xr[c]; s += v; s2 += v * v; }
    const float mean = s / CDIM;
    const float var = s2 / CDIM - mean * mean;
    const float inv = rsqrtf(var + 1e-5f);
    for (int c = 0; c < CDIM; ++c)
      Ax[lane * CDIM + c] = (_Float16)((xr[c] - mean) * inv * gamma[c] + beta[c]);
  }
  __syncthreads();

  const int nlane = lane & 15, hh = lane >> 4;
  // GEMM1: (16 x CDIM) @ (CDIM x 96)
  for (int jn = 0; jn < 6; ++jn) {
    f8 acc = {};
    for (int kc = 0; kc < CDIM / 32; ++kc) {
      if (kc + 1 < CDIM / 32)
        __builtin_prefetch(sqw + (size_t)(kc + 1) * 32 * 96 + jn * 16 + nlane, 0, 3);
      h16 a = load_a(Ax + kc * 32, CDIM, lane);
      h16 b = load_b(sqw + (size_t)kc * 32 * 96 + jn * 16, 96, lane);
      acc = wmma_f16(a, b, acc);
    }
    const int n = jn * 16 + nlane;
#pragma unroll
    for (int v = 0; v < 8; ++v) Y[(v + hh * 8) * 96 + n] = (_Float16)(acc[v] + sqb[n]);
  }
  __syncthreads();

  // GEMM2: (16 x 96) @ (96 x 288) -> scatter into Q/K/V
  const float qscale = 0.10846522890932808f;  // 85^-0.5
  for (int jn = 0; jn < 18; ++jn) {
    f8 acc = {};
    for (int kc = 0; kc < 3; ++kc) {
      h16 a = load_a(Y + kc * 32, 96, lane);
      h16 b = load_b(qkvw + (size_t)kc * 32 * 288 + jn * 16, 288, lane);
      acc = wmma_f16(a, b, acc);
    }
    const int c = jn * 16 + nlane;
    const int g = c / 96, rem = c % 96, h = rem / 12, d = rem % 12;
#pragma unroll
    for (int v = 0; v < 8; ++v) {
      const int tg = tg0 + v + hh * 8;
      const int wi = tg / SQi, t = tg % SQi;
      const float val = acc[v] + qkvb[c];
      const size_t idx = ((size_t)(wi * 8 + h) * 85 + (OFFi + t)) * 12 + d;
      if (g == 0)      Q[idx] = (_Float16)(val * qscale);
      else if (g == 1) K[idx] = (_Float16)val;
      else             V[idx] = (_Float16)val;
    }
  }
}

// ================== Stage 2: 85x85 scale attention per (window, head) ==================
__global__ __launch_bounds__(32)
void attn_scale_kernel(const _Float16* __restrict__ Q, const _Float16* __restrict__ K,
                       const _Float16* __restrict__ V, const float* __restrict__ scale_bias,
                       _Float16* __restrict__ X1) {
  extern __shared__ char smem2[];
  _Float16* Qp = (_Float16*)smem2;                  // 96 x 32
  _Float16* Kp = Qp + 96 * 32;                      // 96 x 32
  _Float16* Vp = Kp + 96 * 32;                      // 96 x 16
  float* attnF = (float*)(Vp + 96 * 16);            // 96 x 96 f32 (raw logits)
  _Float16* attn16 = (_Float16*)(attnF + 96 * 96);  // 96 x 96 f16 (softmaxed)
  float* sbias = (float*)(attn16 + 96 * 96);        // 211 staged bias values
  int* rowOff = (int*)(sbias + 212);                // 96
  int* colOff = rowOff + 96;                        // 96

  const int lane = threadIdx.x;
  const int wi = blockIdx.x >> 3, h = blockIdx.x & 7;
  const size_t base = ((size_t)wi * 8 + h) * 85 * 12;

  // stage bias table + closed-form index tables
  for (int i = lane; i < 211; i += 32) sbias[i] = scale_bias[i * 8 + h];
  for (int i = lane; i < 96; i += 32) {
    const int m = (i < 85) ? i : 84;
    rowOff[i] = m + oh85(seg85(m));
    colOff[i] = (84 - m) + ow85(seg85(m));
  }
  for (int idx = lane; idx < 96 * 32; idx += 32) {
    const int m = idx >> 5, k = idx & 31;
    _Float16 qv = (_Float16)0.f, kv = (_Float16)0.f;
    if (m < 85 && k < 12) { qv = Q[base + m * 12 + k]; kv = K[base + m * 12 + k]; }
    Qp[idx] = qv; Kp[idx] = kv;
  }
  for (int idx = lane; idx < 96 * 16; idx += 32) {
    const int m = idx >> 4, d = idx & 15;
    Vp[idx] = (m < 85 && d < 12) ? V[base + m * 12 + d] : (_Float16)0.f;
  }
  __syncthreads();

  const int nlane = lane & 15, hh = lane >> 4;
  // logits = Qp @ Kp^T  (K-dim padded to 32); branch-free raw stores,
  // bias + masking handled entirely in the softmax pass below.
  for (int mt = 0; mt < 6; ++mt)
    for (int nt = 0; nt < 6; ++nt) {
      f8 acc = {};
      h16 a = load_a(Qp + mt * 16 * 32, 32, lane);
      h16 b = load_bT(Kp + nt * 16 * 32, 32, lane);
      acc = wmma_f16(a, b, acc);
      float* dst = attnF + (mt * 16 + hh * 8) * 96 + nt * 16 + nlane;
#pragma unroll
      for (int v = 0; v < 8; ++v) dst[v * 96] = acc[v];
    }
  __syncthreads();

  // softmax over n<85 per row r<85, absorbing the relative-position bias
  for (int r = lane; r < 96; r += 32) {
    if (r < 85) {
      const int ro = rowOff[r];
      float mx = -3.0e38f;
      for (int n = 0; n < 85; ++n) {
        const float lv = attnF[r * 96 + n] + sbias[ro + colOff[n]];
        attnF[r * 96 + n] = lv;
        mx = fmaxf(mx, lv);
      }
      float sum = 0.f;
      for (int n = 0; n < 85; ++n) sum += __expf(attnF[r * 96 + n] - mx);
      const float inv = 1.f / sum;
      for (int n = 0; n < 96; ++n)
        attn16[r * 96 + n] = (n < 85) ? (_Float16)(__expf(attnF[r * 96 + n] - mx) * inv)
                                      : (_Float16)0.f;
    } else {
      for (int n = 0; n < 96; ++n) attn16[r * 96 + n] = (_Float16)0.f;
    }
  }
  __syncthreads();

  // X1 = attn @ V (K = 96 padded keys, N = 16 padded head dim)
  for (int mt = 0; mt < 6; ++mt) {
    f8 acc = {};
    for (int kc = 0; kc < 3; ++kc) {
      h16 a = load_a(attn16 + mt * 16 * 96 + kc * 32, 96, lane);
      h16 b = load_b(Vp + kc * 32 * 16, 16, lane);
      acc = wmma_f16(a, b, acc);
    }
    const int d = nlane;
#pragma unroll
    for (int v = 0; v < 8; ++v) {
      const int m = mt * 16 + v + hh * 8;
      if (m < 85 && d < 12) X1[base + m * 12 + d] = (_Float16)acc[v];
    }
  }
}

// ================== Stage 3: sqkv = X1 @ sp_w + sp_b (s_q pre-scaled by 1/16) ==================
__global__ void sqkv_kernel(const _Float16* __restrict__ X1, const float* __restrict__ sp_w,
                            const float* __restrict__ sp_b, float* __restrict__ sqkv) {
  const int tid = blockIdx.x * 256 + threadIdx.x;
  if (tid >= 4 * 8 * 256) return;
  const int b = tid >> 11, h = (tid >> 8) & 7, n = tid & 255;
  const int bn = b * 256 + n;
  const _Float16* xr = X1 + ((size_t)bn * 8 + h) * 1020;
  float a0 = 0.f, a1 = 0.f, a2 = 0.f;
  for (int f = 0; f < 1020; ++f) {
    const float xv = (float)xr[f];
    a0 += xv * sp_w[f * 3 + 0];
    a1 += xv * sp_w[f * 3 + 1];
    a2 += xv * sp_w[f * 3 + 2];
  }
  const size_t o = ((size_t)(b * 8 + h) * 256 + n) * 3;
  sqkv[o + 0] = (a0 + sp_b[0]) * 0.0625f;  // n^-0.5 = 1/16
  sqkv[o + 1] = a1 + sp_b[1];
  sqkv[o + 2] = a2 + sp_b[2];
}

// ================== Stage 4: spatial attention, per (b, h, 16-row tile) ==================
__global__ __launch_bounds__(32)
void attn_spatial_kernel(const _Float16* __restrict__ X1, const float* __restrict__ sqkv,
                         const float* __restrict__ spatial_bias, _Float16* __restrict__ X2) {
  __shared__ float rowsF[16 * 256];
  __shared__ _Float16 rows16[16 * 256];
  __shared__ float sv[256];
  __shared__ float sb[961];
  const int lane = threadIdx.x;
  const int mt = blockIdx.x & 15;
  const int h = (blockIdx.x >> 4) & 7;
  const int b = blockIdx.x >> 7;
  const float* sq_base = sqkv + (size_t)(b * 8 + h) * 256 * 3;

  for (int m = lane; m < 256; m += 32) sv[m] = sq_base[m * 3 + 2];
  for (int i = lane; i < 961; i += 32) sb[i] = spatial_bias[i * 8 + h];
  __syncthreads();
  for (int idx = lane; idx < 16 * 256; idx += 32) {
    const int r = idx >> 8, m = idx & 255;
    const int n = mt * 16 + r;
    const float s_q = sq_base[n * 3 + 0];
    const float s_k = sq_base[m * 3 + 1];
    const int i = n >> 4, j = n & 15, k2 = m >> 4, l = m & 15;
    const int sidx = (i - k2 + 15) * 31 + (j - l + 15);
    rowsF[idx] = s_q * s_k + sb[sidx];
  }
  __syncthreads();
  if (lane < 16) {
    const int r = lane;
    float mx = -3.0e38f;
    for (int m = 0; m < 256; ++m) mx = fmaxf(mx, rowsF[r * 256 + m]);
    float sum = 0.f;
    for (int m = 0; m < 256; ++m) sum += __expf(rowsF[r * 256 + m] - mx);
    const float inv = 1.f / sum;
    for (int m = 0; m < 256; ++m)
      rows16[r * 256 + m] = (_Float16)(__expf(rowsF[r * 256 + m] - mx) * inv);
  }
  __syncthreads();

  const int nlane = lane & 15, hh = lane >> 4;
  for (int nt = 0; nt < 64; ++nt) {  // N = 1020 padded to 1024
    f8 acc = {};
    for (int kc = 0; kc < 8; ++kc) {  // K = 256
      h16 a = load_a(rows16 + kc * 32, 256, lane);
      h16 bfrag;
      const int f0 = nt * 16 + nlane;
#pragma unroll
      for (int j = 0; j < 8; ++j) {
        const int k = hh * 16 + 2 * j;
#pragma unroll
        for (int p = 0; p < 2; ++p) {
          const int m = kc * 32 + k + p;
          float val = 0.f;
          if (f0 < 1020)
            val = sv[m] * (float)X1[((size_t)(b * 256 + m) * 8 + h) * 1020 + f0];
          bfrag[2 * j + p] = (_Float16)val;
        }
      }
      acc = wmma_f16(a, bfrag, acc);
    }
#pragma unroll
    for (int v = 0; v < 8; ++v) {
      const int nrow = mt * 16 + v + hh * 8;
      const int f = nt * 16 + nlane;
      if (f < 1020) {
        const int s = f / 12, d = f % 12;
        X2[((size_t)(b * 256 + nrow) * 85 + s) * 96 + h * 12 + d] = (_Float16)acc[v];
      }
    }
  }
}

// ================== Stage 5: window reverse + LN + rv proj + MLP + residual ==================
template<int CDIM, int SQi, int WSi, int OFFi>
__global__ __launch_bounds__(32)
void tail_kernel(const _Float16* __restrict__ X2,
                 const float* __restrict__ gamma, const float* __restrict__ beta,
                 const _Float16* __restrict__ rvw, const float* __restrict__ rvb,
                 const _Float16* __restrict__ w1, const float* __restrict__ b1,
                 const _Float16* __restrict__ w2, const float* __restrict__ b2,
                 float* __restrict__ out) {
  extern __shared__ char smem5[];
  _Float16* Aln = (_Float16*)smem5;                  // 16 x 96 f16
  float* outAcc = (float*)(Aln + 16 * 96);           // 16 x CDIM f32 (proj + mlp accum)
  _Float16* proj16 = (_Float16*)(outAcc + 16 * CDIM);// 16 x CDIM f16 (mlp1 input)
  _Float16* hid = proj16 + 16 * CDIM;                // 16 x 32 f16 hidden chunk
  const int H4 = 4 * CDIM;
  const int lane = threadIdx.x;
  const int tg0 = blockIdx.x * 16;
  const int HW = WSi * 16;

  if (lane < 16) {
    const int tg = tg0 + lane;
    const int wi = tg / SQi, t = tg % SQi;
    const _Float16* xr = X2 + ((size_t)wi * 85 + OFFi + t) * 96;
    float s = 0.f, s2 = 0.f;
    for (int c = 0; c < 96; ++c) { float v = (float)xr[c]; s += v; s2 += v * v; }
    const float mean = s * (1.f / 96.f);
    const float var = s2 * (1.f / 96.f) - mean * mean;
    const float inv = rsqrtf(var + 1e-5f);
    for (int c = 0; c < 96; ++c)
      Aln[lane * 96 + c] = (_Float16)(((float)xr[c] - mean) * inv * gamma[c] + beta[c]);
  }
  __syncthreads();

  const int nlane = lane & 15, hh = lane >> 4;
  // proj = Aln @ rvw + rvb -> outAcc (f32) and proj16 (f16)
  for (int jt = 0; jt < CDIM / 16; ++jt) {
    f8 acc = {};
    for (int kc = 0; kc < 3; ++kc) {
      h16 a = load_a(Aln + kc * 32, 96, lane);
      h16 b = load_b(rvw + (size_t)kc * 32 * CDIM + jt * 16, CDIM, lane);
      acc = wmma_f16(a, b, acc);
    }
    const int c = jt * 16 + nlane;
#pragma unroll
    for (int v = 0; v < 8; ++v) {
      const int m = v + hh * 8;
      const float val = acc[v] + rvb[c];
      outAcc[m * CDIM + c] = val;
      proj16[m * CDIM + c] = (_Float16)val;
    }
  }
  __syncthreads();

  // stream hidden in chunks of 32, accumulate mlp2 into outAcc
  for (int hc = 0; hc < H4 / 32; ++hc) {
    for (int sub = 0; sub < 2; ++sub) {
      f8 acc = {};
      for (int kc = 0; kc < CDIM / 32; ++kc) {
        if (kc + 1 < CDIM / 32)
          __builtin_prefetch(w1 + (size_t)(kc + 1) * 32 * H4 + hc * 32 + sub * 16 + nlane, 0, 3);
        h16 a = load_a(proj16 + kc * 32, CDIM, lane);
        h16 b = load_b(w1 + (size_t)kc * 32 * H4 + hc * 32 + sub * 16, H4, lane);
        acc = wmma_f16(a, b, acc);
      }
      const int cl = sub * 16 + nlane;
      const int cg = hc * 32 + cl;
#pragma unroll
      for (int v = 0; v < 8; ++v) {
        const float hv = acc[v] + b1[cg];
        const float g = 0.5f * hv * (1.f + erff(hv * 0.70710678118654752f));  // exact GELU
        hid[(v + hh * 8) * 32 + cl] = (_Float16)g;
      }
    }
    __syncthreads();
    for (int jt = 0; jt < CDIM / 16; ++jt) {
      const int c = jt * 16 + nlane;
      f8 cacc;
#pragma unroll
      for (int v = 0; v < 8; ++v) cacc[v] = outAcc[(v + hh * 8) * CDIM + c];
      h16 a = load_a(hid, 32, lane);
      h16 b = load_b(w2 + (size_t)hc * 32 * CDIM + jt * 16, CDIM, lane);
      cacc = wmma_f16(a, b, cacc);
#pragma unroll
      for (int v = 0; v < 8; ++v) outAcc[(v + hh * 8) * CDIM + c] = cacc[v];
    }
    __syncthreads();
  }

  // out = proj + h (b2 added here), NCHW scatter
  for (int idx = lane; idx < 16 * CDIM; idx += 32) {
    const int r = idx / CDIM, c = idx % CDIM;
    const int tg = tg0 + r;
    const int wi = tg / SQi, t = tg % SQi;
    const int b = wi >> 8, wy = (wi >> 4) & 15, wx = wi & 15;
    const int ty = t / WSi, tx = t % WSi;
    const int row = wy * WSi + ty, col = wx * WSi + tx;
    out[((size_t)(b * CDIM + c) * HW + row) * HW + col] = outAcc[idx] + b2[c];
  }
}

// ================== host launcher ==================
extern "C" void kernel_launch(void* const* d_in, const int* in_sizes, int n_in,
                              void* d_out, int out_size, void* d_ws, size_t ws_size,
                              hipStream_t stream) {
  (void)in_sizes; (void)n_in; (void)out_size; (void)ws_size;
  const int DIMS[4] = {96, 192, 384, 768};
  auto F = [&](int i) { return (const float*)d_in[i]; };
  // input order: x0..x3, then 4-lists: sq_gamma, sq_beta, sq_w, sq_b, qkv_w, qkv_b,
  // rv_gamma, rv_beta, rv_w, rv_b, mlp_w1, mlp_b1, mlp_w2, mlp_b2, then sp_w, sp_b,
  // scale_bias, spatial_bias  (64 inputs total)

  char* ws = (char*)d_ws;
  size_t o = 0;  // offset in halves
  auto alloc_h = [&](size_t n) { _Float16* p = (_Float16*)(ws + o * 2); o += n; return p; };
  _Float16 *sqw16[4], *qkvw16[4], *rvw16[4], *w116[4], *w216[4];
  for (int i = 0; i < 4; ++i) sqw16[i]  = alloc_h((size_t)DIMS[i] * 96);
  for (int i = 0; i < 4; ++i) qkvw16[i] = alloc_h((size_t)96 * 288);
  for (int i = 0; i < 4; ++i) rvw16[i]  = alloc_h((size_t)96 * DIMS[i]);
  for (int i = 0; i < 4; ++i) w116[i]   = alloc_h((size_t)DIMS[i] * 4 * DIMS[i]);
  for (int i = 0; i < 4; ++i) w216[i]   = alloc_h((size_t)DIMS[i] * 4 * DIMS[i]);
  _Float16* Q  = alloc_h((size_t)1024 * 8 * 85 * 12);
  _Float16* K  = alloc_h((size_t)1024 * 8 * 85 * 12);
  _Float16* V  = alloc_h((size_t)1024 * 8 * 85 * 12);
  _Float16* X1 = alloc_h((size_t)1024 * 8 * 85 * 12);
  _Float16* X2 = alloc_h((size_t)1024 * 85 * 96);
  o = (o + 7) & ~(size_t)7;
  float* sqkv = (float*)(ws + o * 2);  // 4*8*256*3 floats

  auto cvt = [&](const float* s, _Float16* d, size_t n) {
    cvt_f32_f16<<<(unsigned)((n + 255) / 256), 256, 0, stream>>>(s, d, (int)n);
  };
  for (int i = 0; i < 4; ++i) {
    cvt(F(12 + i), sqw16[i],  (size_t)DIMS[i] * 96);
    cvt(F(20 + i), qkvw16[i], (size_t)96 * 288);
    cvt(F(36 + i), rvw16[i],  (size_t)96 * DIMS[i]);
    cvt(F(44 + i), w116[i],   (size_t)DIMS[i] * 4 * DIMS[i]);
    cvt(F(52 + i), w216[i],   (size_t)DIMS[i] * 4 * DIMS[i]);
  }

  // Stage 1: per-scale QKV
  qkv_kernel<96, 64, 8, 0><<<65536 / 16, 32, 0, stream>>>(
      F(0), F(4), F(8), sqw16[0], F(16), qkvw16[0], F(24), Q, K, V);
  qkv_kernel<192, 16, 4, 64><<<16384 / 16, 32, 0, stream>>>(
      F(1), F(5), F(9), sqw16[1], F(17), qkvw16[1], F(25), Q, K, V);
  qkv_kernel<384, 4, 2, 80><<<4096 / 16, 32, 0, stream>>>(
      F(2), F(6), F(10), sqw16[2], F(18), qkvw16[2], F(26), Q, K, V);
  qkv_kernel<768, 1, 1, 84><<<1024 / 16, 32, 0, stream>>>(
      F(3), F(7), F(11), sqw16[3], F(19), qkvw16[3], F(27), Q, K, V);

  // Stage 2: scale attention (dyn LDS: Qp/Kp/Vp + attnF + attn16 + sbias + idx tables)
  const int smem2_sz = 96*32*2*2 + 96*16*2 + 96*96*4 + 96*96*2 + 212*4 + 2*96*4;
  attn_scale_kernel<<<1024 * 8, 32, smem2_sz, stream>>>(Q, K, V, F(62), X1);

  // Stage 3: sqkv projection
  sqkv_kernel<<<8192 / 256, 256, 0, stream>>>(X1, F(60), F(61), sqkv);

  // Stage 4: spatial attention
  attn_spatial_kernel<<<512, 32, 0, stream>>>(X1, sqkv, F(63), X2);

  // Stage 5: per-scale tail (dyn LDS = 4096 + 96*CDIM bytes)
  float* out = (float*)d_out;
  tail_kernel<96, 64, 8, 0><<<65536 / 16, 32, 4096 + 96 * 96, stream>>>(
      X2, F(28), F(32), rvw16[0], F(40), w116[0], F(48), w216[0], F(56), out + 0);
  tail_kernel<192, 16, 4, 64><<<16384 / 16, 32, 4096 + 96 * 192, stream>>>(
      X2, F(29), F(33), rvw16[1], F(41), w116[1], F(49), w216[1], F(57), out + 6291456);
  tail_kernel<384, 4, 2, 80><<<4096 / 16, 32, 4096 + 96 * 384, stream>>>(
      X2, F(30), F(34), rvw16[2], F(42), w116[2], F(50), w216[2], F(58), out + 9437184);
  tail_kernel<768, 1, 1, 84><<<1024 / 16, 32, 4096 + 96 * 768, stream>>>(
      X2, F(31), F(35), rvw16[3], F(43), w116[3], F(51), w216[3], F(59), out + 11010048);
}
